// MyCustomAttentionMap_73538430042490
// MI455X (gfx1250) — matrix operand order repaired and checked by reading, not compiled, not run on hardware
//
#include <hip/hip_runtime.h>
#include <hip/hip_bf16.h>

// ---------------------------------------------------------------------------
// Types for CDNA5 WMMA (wave32): 16x16x32 bf16 -> f32
// ---------------------------------------------------------------------------
typedef __attribute__((ext_vector_type(16))) __bf16 v16bf;
typedef __attribute__((ext_vector_type(8)))  __bf16 v8bf;
typedef __attribute__((ext_vector_type(4)))  __bf16 v4bf;
typedef __attribute__((ext_vector_type(8)))  float  v8f;

#define BN   8
#define SEQ  1024
#define CH   768
#define NH   12
#define HD   64
#define MT   (BN * SEQ)   // 8192 rows
#define OUT3 (3 * CH)     // 2304

// ---------------------------------------------------------------------------
// Fragment loaders matching the CDNA5 ISA 16-bit WMMA VGPR layouts.
// A (16x32, MxK): lane m (0-15) holds K={h*8..h*8+7, 16+h*8..16+h*8+7},
//                 where h = lane>>4. Two 16B contiguous chunks per lane.
// B (32x16, KxN): lane n (0-15) column n, same K order as A:
//                 K={h*8..h*8+7, 16+h*8..16+h*8+7} (source stored
//                 "N rows x K contiguous", e.g. W[o][c]).
// ---------------------------------------------------------------------------
__device__ __forceinline__ v16bf load_a_frag(const __bf16* base, int ld) {
  const int lane = threadIdx.x & 31;
  const int m = lane & 15, hh = lane >> 4;
  const __bf16* p = base + (size_t)m * ld + hh * 8;
  union { v16bf v; v8bf c[2]; } u;
  u.c[0] = *(const v8bf*)(p);
  u.c[1] = *(const v8bf*)(p + 16);
  return u.v;
}

__device__ __forceinline__ v16bf load_b_frag(const __bf16* base, int ld) {
  const int lane = threadIdx.x & 31;
  const int n = lane & 15, hh = lane >> 4;
  const __bf16* p = base + (size_t)n * ld + hh * 8;
  union { v16bf v; v8bf c[2]; } u;
  u.c[0] = *(const v8bf*)(p);
  u.c[1] = *(const v8bf*)(p + 16);
  return u.v;
}

__device__ __forceinline__ v8f wmma_bf16(v16bf a, v16bf b, v8f c) {
  // (neg_a, A, neg_b, B, c_mod, C, reuse_a, reuse_b)
  return __builtin_amdgcn_wmma_f32_16x16x32_bf16(false, a, false, b,
                                                 (short)0, c, false, false);
}

// ---------------------------------------------------------------------------
// CDNA5 async global->LDS copy (16B per lane) + completion wait.
// VDST = per-lane LDS byte offset, VADDR = 64-bit global address, no SADDR.
// Tracked by ASYNCcnt (CDNA5 ISA).
// ---------------------------------------------------------------------------
__device__ __forceinline__ void async_copy_b128(unsigned lds_off,
                                                const void* gaddr) {
  asm volatile("global_load_async_to_lds_b128 %0, %1, off"
               :: "v"(lds_off), "v"(gaddr) : "memory");
}
__device__ __forceinline__ void wait_asynccnt0() {
  asm volatile("s_wait_asynccnt 0x0" ::: "memory");
}

// ---------------------------------------------------------------------------
// fp32 -> bf16 conversion (vectorized x4)
// ---------------------------------------------------------------------------
__global__ __launch_bounds__(256) void cvt_f32_bf16(
    const float* __restrict__ src, __bf16* __restrict__ dst, int n4) {
  int i = blockIdx.x * 256 + threadIdx.x;
  if (i < n4) {
    float4 f = ((const float4*)src)[i];
    v4bf o;
    o[0] = (__bf16)f.x; o[1] = (__bf16)f.y;
    o[2] = (__bf16)f.z; o[3] = (__bf16)f.w;
    ((v4bf*)dst)[i] = o;
  }
}

// ---------------------------------------------------------------------------
// One-time V transpose: [B,H,N,D] -> [B,H,D,N] (bf16), so the flash loop can
// stage V^T chunks with pure 16B async copies (no in-loop transpose).
// ---------------------------------------------------------------------------
__global__ __launch_bounds__(256) void transpose_v(
    const __bf16* __restrict__ v, __bf16* __restrict__ vt) {
  const int bh = blockIdx.y;
  const int n  = blockIdx.x * 32 + (threadIdx.x >> 3);
  const int d0 = (threadIdx.x & 7) * 8;
  v8bf vv = *(const v8bf*)(v + ((size_t)bh * SEQ + n) * HD + d0);
  __bf16* dst = vt + (size_t)bh * HD * SEQ;
  for (int j = 0; j < 8; ++j) dst[(size_t)(d0 + j) * SEQ + n] = vv[j];
}

// ---------------------------------------------------------------------------
// Shared GEMM core: block tile 128(M) x 128(N), 8 waves in a 4x2 grid, each
// wave owns a 32x64 tile = 2x4 WMMA accumulators (8 WMMA per 32-K step, only
// 6 fragment loads -> 0.75 b128-pairs per WMMA). Fragments come straight from
// global in WMMA lane layout; global_prefetch_b8 pulls K two steps ahead.
// ---------------------------------------------------------------------------
template <typename EPI>
__device__ __forceinline__ void gemm_core_128x128(
    const __bf16* __restrict__ A, const __bf16* __restrict__ Bw,
    int m0, int o0, EPI epilogue) {
  const int wid = threadIdx.x >> 5;
  const int wm  = wid & 3;       // 4 row-groups of 32
  const int wn  = wid >> 2;      // 2 col-groups of 64
  const int lane = threadIdx.x & 31;

  const __bf16* arow = A  + (size_t)(m0 + wm * 32) * CH;
  const __bf16* brow = Bw + (size_t)(o0 + wn * 64) * CH;

  v8f acc[2][4] = {};
  for (int kk = 0; kk < CH; kk += 32) {
    if (kk + 64 < CH) {  // speculative prefetch two steps ahead
      __builtin_prefetch(arow + (size_t)(lane & 15) * CH + kk + 64, 0, 1);
      __builtin_prefetch(brow + (size_t)(lane & 15) * CH + kk + 64, 0, 1);
    }
    v16bf a0 = load_a_frag(arow + kk, CH);
    v16bf a1 = load_a_frag(arow + (size_t)16 * CH + kk, CH);
    v16bf b0 = load_b_frag(brow + kk, CH);
    v16bf b1 = load_b_frag(brow + (size_t)16 * CH + kk, CH);
    v16bf b2 = load_b_frag(brow + (size_t)32 * CH + kk, CH);
    v16bf b3 = load_b_frag(brow + (size_t)48 * CH + kk, CH);
    acc[0][0] = wmma_bf16(a0, b0, acc[0][0]);
    acc[0][1] = wmma_bf16(a0, b1, acc[0][1]);
    acc[0][2] = wmma_bf16(a0, b2, acc[0][2]);
    acc[0][3] = wmma_bf16(a0, b3, acc[0][3]);
    acc[1][0] = wmma_bf16(a1, b0, acc[1][0]);
    acc[1][1] = wmma_bf16(a1, b1, acc[1][1]);
    acc[1][2] = wmma_bf16(a1, b2, acc[1][2]);
    acc[1][3] = wmma_bf16(a1, b3, acc[1][3]);
  }

  const int colb = lane & 15, rowh = (lane >> 4) * 8;
  for (int ti = 0; ti < 2; ++ti)
    for (int tj = 0; tj < 4; ++tj)
      for (int r = 0; r < 8; ++r) {
        int m = m0 + wm * 32 + ti * 16 + rowh + r;
        int o = o0 + wn * 64 + tj * 16 + colb;
        epilogue(m, o, acc[ti][tj][r]);
      }
}

// ---------------------------------------------------------------------------
// QKV GEMM: out[m,o] = sum_c x[m,c] * w[o,c] + bias[o]
// Epilogue scatters into q/k/v laid out [B,H,N,D] (bf16).
// ---------------------------------------------------------------------------
__global__ __launch_bounds__(256) void qkv_gemm(
    const __bf16* __restrict__ A,      // [8192][768] x in bf16
    const __bf16* __restrict__ Bw,     // [2304][768] qkv_w in bf16
    const float*  __restrict__ bias,   // [2304]
    __bf16* __restrict__ q, __bf16* __restrict__ k, __bf16* __restrict__ v) {
  gemm_core_128x128(A, Bw, blockIdx.x * 128, blockIdx.y * 128,
    [&](int m, int o, float accv) {
      float val = accv + bias[o];
      int b = m >> 10, n = m & 1023;
      int sec = o / CH, w_in = o % CH;
      int hh = w_in >> 6, d = w_in & 63;
      __bf16* dstp = (sec == 0) ? q : (sec == 1) ? k : v;
      dstp[(((size_t)b * NH + hh) * SEQ + n) * HD + d] = (__bf16)val;
    });
}

// ---------------------------------------------------------------------------
// Projection GEMM: out[m,o] = sum_c A[m,c] * w[o,c] + bias[o] (fp32 out)
// ---------------------------------------------------------------------------
__global__ __launch_bounds__(256) void proj_gemm(
    const __bf16* __restrict__ A,      // [8192][768] attn out bf16
    const __bf16* __restrict__ Bw,     // [768][768] proj_w bf16
    const float*  __restrict__ bias,   // [768]
    float* __restrict__ out) {         // [8192][768]
  gemm_core_128x128(A, Bw, blockIdx.x * 128, blockIdx.y * 128,
    [&](int m, int o, float accv) {
      out[(size_t)m * CH + o] = accv + bias[o];
    });
}

// ---------------------------------------------------------------------------
// LayerNorm over D=64 for q (x scale) and k, in-place on bf16 [B,H,N,D].
// 8 lanes per row, shuffle-tree reduction (wave32).
// ---------------------------------------------------------------------------
__global__ __launch_bounds__(256) void ln_qk(
    __bf16* __restrict__ q, __bf16* __restrict__ k,
    const float* __restrict__ qg, const float* __restrict__ qb2,
    const float* __restrict__ kg, const float* __restrict__ kb2) {
  const int tid = threadIdx.x;
  const int rl = tid >> 3, l8 = tid & 7;
  size_t gr = (size_t)blockIdx.x * 32 + rl;
  const size_t NR = (size_t)BN * NH * SEQ;  // 98304 rows each
  __bf16* ptr; const float *g, *bb; float scale;
  if (gr < NR) { ptr = q; g = qg; bb = qb2; scale = 0.125f; }  // 64^-0.5
  else         { gr -= NR; ptr = k; g = kg; bb = kb2; scale = 1.0f; }

  __bf16* row = ptr + gr * HD + l8 * 8;
  v8bf xv = *(const v8bf*)row;
  float xf[8], s = 0.f, s2 = 0.f;
  for (int j = 0; j < 8; ++j) {
    xf[j] = (float)xv[j];
    s += xf[j]; s2 += xf[j] * xf[j];
  }
  for (int m = 1; m < 8; m <<= 1) {
    s  += __shfl_xor(s,  m, 32);
    s2 += __shfl_xor(s2, m, 32);
  }
  float mu  = s * (1.0f / 64.0f);
  float var = s2 * (1.0f / 64.0f) - mu * mu;
  float rs  = rsqrtf(var + 1e-5f);
  v8bf ov;
  for (int j = 0; j < 8; ++j) {
    int d = l8 * 8 + j;
    float y = (xf[j] - mu) * rs * g[d] + bb[d];
    ov[j] = (__bf16)(y * scale);
  }
  *(v8bf*)row = ov;
}

// ---------------------------------------------------------------------------
// Flash-style attention: block = 128 query rows (8 waves x 16 rows), loops
// over keys in chunks of 32. Per iteration:
//   1) kick async global->LDS copy of the V^T chunk (64 d-rows x 32 keys)
//   2) S via 4 WMMA + online softmax (overlapped with the async copy)
//   3) s_wait_asynccnt 0 + barrier, then PV via 4 WMMA from LDS
// P is re-laid-out through LDS into an A fragment. Output -> [B,N,C] bf16.
// ---------------------------------------------------------------------------
__global__ __launch_bounds__(256) void attn_kernel(
    const __bf16* __restrict__ q, const __bf16* __restrict__ k,
    const __bf16* __restrict__ vt, __bf16* __restrict__ out) {
  __shared__ __attribute__((aligned(16))) __bf16 v_t[HD][32];       // 4 KB
  __shared__ __attribute__((aligned(16))) __bf16 p_lds[8][16][32];  // 8 KB

  const int bh = blockIdx.y;            // 0..95
  const int b = bh / NH, h = bh % NH;
  const int wid  = threadIdx.x >> 5;
  const int lane = threadIdx.x & 31;
  const int colb = lane & 15, half = lane >> 4;

  const __bf16* qp  = q  + (size_t)bh * SEQ * HD;
  const __bf16* kp  = k  + (size_t)bh * SEQ * HD;
  const __bf16* vtp = vt + (size_t)bh * HD * SEQ;   // [64][1024]

  const int q0 = blockIdx.x * 128 + wid * 16;
  v16bf aq0 = load_a_frag(qp + (size_t)q0 * HD, HD);       // d 0..31
  v16bf aq1 = load_a_frag(qp + (size_t)q0 * HD + 32, HD);  // d 32..63

  v8f o_acc[4] = {};
  float m_run[8], l_run[8];
  for (int r = 0; r < 8; ++r) { m_run[r] = -3.0e38f; l_run[r] = 0.0f; }

  // async staging geometry: one b128 per thread, 64 rows x 64B
  const int srow = threadIdx.x >> 2;        // d row 0..63
  const int sseg = (threadIdx.x & 3) * 8;   // element offset within chunk row
  const unsigned v_t_base =
      (unsigned)(size_t)(&v_t[0][0]) + (unsigned)((srow * 32 + sseg) * 2);
  const __bf16* vrow = vtp + (size_t)srow * SEQ + sseg;

  for (int kb = 0; kb < SEQ; kb += 32) {
    __syncthreads();  // previous PV reads of v_t complete
    // 1) async stage V^T chunk (overlaps with S + softmax below)
    async_copy_b128(v_t_base, vrow + kb);

    if (kb + 32 < SEQ) {  // prefetch next K/V^T chunk
      __builtin_prefetch(kp + (size_t)(kb + 32 + lane) * HD, 0, 1);
      __builtin_prefetch(vtp + (size_t)lane * SEQ + kb + 32, 0, 1);
      __builtin_prefetch(vtp + (size_t)(lane + 32) * SEQ + kb + 32, 0, 1);
    }

    // 2) S tiles (16 q-rows x 32 keys), K depth 64 = 2 chained WMMA each
    v8f s0 = {}, s1 = {};
    {
      v16bf bk;
      bk = load_b_frag(kp + (size_t)kb * HD, HD);             s0 = wmma_bf16(aq0, bk, s0);
      bk = load_b_frag(kp + (size_t)kb * HD + 32, HD);        s0 = wmma_bf16(aq1, bk, s0);
      bk = load_b_frag(kp + (size_t)(kb + 16) * HD, HD);      s1 = wmma_bf16(aq0, bk, s1);
      bk = load_b_frag(kp + (size_t)(kb + 16) * HD + 32, HD); s1 = wmma_bf16(aq1, bk, s1);
    }

    // online softmax per row (row lives in a 16-lane half-group)
    for (int r = 0; r < 8; ++r) {
      float mx = fmaxf(s0[r], s1[r]);
      for (int mk = 1; mk < 16; mk <<= 1) mx = fmaxf(mx, __shfl_xor(mx, mk, 32));
      float m_new = fmaxf(m_run[r], mx);
      float corr = __expf(m_run[r] - m_new);
      float p0 = __expf(s0[r] - m_new);
      float p1 = __expf(s1[r] - m_new);
      float ps = p0 + p1;
      for (int mk = 1; mk < 16; mk <<= 1) ps += __shfl_xor(ps, mk, 32);
      l_run[r] = l_run[r] * corr + ps;
      m_run[r] = m_new;
      for (int t = 0; t < 4; ++t) o_acc[t][r] *= corr;
      p_lds[wid][half * 8 + r][colb]      = (__bf16)p0;
      p_lds[wid][half * 8 + r][colb + 16] = (__bf16)p1;
    }

    // 3) V^T chunk landed + P visible to the whole wavefront group
    wait_asynccnt0();
    __syncthreads();

    // O += P @ V  (A = 16x32 P, B = 32x16 slices of V^T)
    v16bf ap = load_a_frag(&p_lds[wid][0][0], 32);
    for (int t = 0; t < 4; ++t) {
      v16bf bv = load_b_frag(&v_t[t * 16][0], 32);
      o_acc[t] = wmma_bf16(ap, bv, o_acc[t]);
    }
  }

  // epilogue -> [B,N,C] bf16 (heads re-interleaved)
  for (int t = 0; t < 4; ++t)
    for (int r = 0; r < 8; ++r) {
      int qrow = q0 + half * 8 + r;
      int d = t * 16 + colb;
      float val = o_acc[t][r] / l_run[r];
      out[((size_t)(b * SEQ + qrow)) * CH + h * HD + d] = (__bf16)val;
    }
}

// ---------------------------------------------------------------------------
// Host launch
// ---------------------------------------------------------------------------
extern "C" void kernel_launch(void* const* d_in, const int* in_sizes, int n_in,
                              void* d_out, int out_size, void* d_ws, size_t ws_size,
                              hipStream_t stream) {
  const float* x      = (const float*)d_in[0];
  const float* qkv_w  = (const float*)d_in[1];
  const float* qkv_b  = (const float*)d_in[2];
  const float* qn_g   = (const float*)d_in[3];
  const float* qn_b   = (const float*)d_in[4];
  const float* kn_g   = (const float*)d_in[5];
  const float* kn_b   = (const float*)d_in[6];
  const float* proj_w = (const float*)d_in[7];
  const float* proj_b = (const float*)d_in[8];

  char* ws = (char*)d_ws;
  size_t off = 0;
  auto carve = [&](size_t bytes) {
    char* p = ws + off;
    off += (bytes + 255) & ~(size_t)255;
    return p;
  };
  __bf16* x_bf   = (__bf16*)carve((size_t)MT * CH * 2);    // 12.6 MB
  __bf16* wqkv   = (__bf16*)carve((size_t)OUT3 * CH * 2);  //  3.5 MB
  __bf16* wproj  = (__bf16*)carve((size_t)CH * CH * 2);    //  1.2 MB
  __bf16* qbuf   = (__bf16*)carve((size_t)MT * CH * 2);    // 12.6 MB
  __bf16* kbuf   = (__bf16*)carve((size_t)MT * CH * 2);    // 12.6 MB
  __bf16* vbuf   = (__bf16*)carve((size_t)MT * CH * 2);    // 12.6 MB
  __bf16* vtbuf  = (__bf16*)carve((size_t)MT * CH * 2);    // 12.6 MB (V^T)
  __bf16* attn   = (__bf16*)carve((size_t)MT * CH * 2);    // 12.6 MB

  int n4;
  n4 = MT * CH / 4;
  cvt_f32_bf16<<<(n4 + 255) / 256, 256, 0, stream>>>(x, x_bf, n4);
  n4 = OUT3 * CH / 4;
  cvt_f32_bf16<<<(n4 + 255) / 256, 256, 0, stream>>>(qkv_w, wqkv, n4);
  n4 = CH * CH / 4;
  cvt_f32_bf16<<<(n4 + 255) / 256, 256, 0, stream>>>(proj_w, wproj, n4);

  qkv_gemm<<<dim3(MT / 128, OUT3 / 128), 256, 0, stream>>>(
      x_bf, wqkv, qkv_b, qbuf, kbuf, vbuf);

  transpose_v<<<dim3(SEQ / 32, BN * NH), 256, 0, stream>>>(vbuf, vtbuf);

  ln_qk<<<(2 * BN * NH * SEQ) / 32, 256, 0, stream>>>(
      qbuf, kbuf, qn_g, qn_b, kn_g, kn_b);

  attn_kernel<<<dim3(SEQ / 128, BN * NH), 256, 0, stream>>>(
      qbuf, kbuf, vtbuf, attn);

  proj_gemm<<<dim3(MT / 128, CH / 128), 256, 0, stream>>>(
      attn, wproj, proj_b, (float*)d_out);
}
